// QuantLinearA16W4_57174604644898
// MI455X (gfx1250) — matrix operand order repaired and checked by reading, not compiled, not run on hardware
//
#include <hip/hip_runtime.h>

#define IN_F   4096
#define OUT_F  11008
#define GS     128
#define BM     128
#define BN     128
#define BK     32
#define LDSS   56            // LDS row stride in bf16 elements (112 B, 16B-aligned, conflict-friendly)
#define NKS    (IN_F / BK)   // 128 K-steps

typedef __attribute__((ext_vector_type(16))) __bf16       v16bf;
typedef __attribute__((ext_vector_type(2)))  __bf16       v2bf;
typedef __attribute__((ext_vector_type(8)))  float        v8f;
typedef __attribute__((ext_vector_type(4)))  float        v4f;
typedef __attribute__((ext_vector_type(4)))  int          v4i;
typedef __attribute__((ext_vector_type(4)))  unsigned int v4u;

union Frag { v16bf v; v4u q[2]; };

// f32 pair -> packed bf16 (a -> low 16, b -> high 16)
#if __has_builtin(__builtin_amdgcn_cvt_pk_bf16_f32)
__device__ __forceinline__ unsigned int pk_bf16(float a, float b) {
  v2bf r = __builtin_amdgcn_cvt_pk_bf16_f32(a, b);   // 1x v_cvt_pk_bf16_f32, RNE
  return __builtin_bit_cast(unsigned int, r);
}
#else
__device__ __forceinline__ unsigned int pk_bf16(float a, float b) {
  unsigned int ua = __float_as_uint(a) + 0x8000u;    // round-to-nearest
  unsigned int ub = __float_as_uint(b) + 0x8000u;
  return __builtin_amdgcn_perm(ub, ua, 0x07060302u); // {ub_hi16, ua_hi16}
}
#endif

__global__ __launch_bounds__(256, 1)
void QuantLinearA16W4_57174604644898_kernel(const float* __restrict__ x,
                                            const int*   __restrict__ qw,
                                            const float* __restrict__ zeros,
                                            const float* __restrict__ scales,
                                            const float* __restrict__ bias,
                                            float*       __restrict__ out) {
  __shared__ alignas(16) unsigned short sA[2][BM * LDSS];  // x tile, bf16
  __shared__ alignas(16) unsigned short sB[2][BN * LDSS];  // dequantized W tile, bf16

  const int tid    = threadIdx.x;
  const int lane   = tid & 31;
  const int wave   = tid >> 5;
  const int wave_m = wave >> 2;   // 0..1  -> 64 rows each
  const int wave_n = wave & 3;    // 0..3  -> 32 cols each

  // Staging: 2 threads per row, each covers 16 K-values (= 8 int32 nibble-pairs / 16 floats)
  const int    srow  = tid >> 1;
  const int    shalf = tid & 1;
  const size_t xrow  = (size_t)blockIdx.y * BM + srow;
  const size_t orow  = (size_t)blockIdx.x * BN + srow;
  const float* xbase = x + xrow * IN_F;
  const int*   qbase = qw + ((orow * IN_F) >> 1);
  const int    gbase = (int)orow * (IN_F / GS);

  v4f   gx[4];
  v4i   gq[2];
  float gsc, gnzs;   // scale, and -(zero+1024)*scale

  auto LOADG = [&](int kb) {
    const v4f* xp = (const v4f*)(xbase + kb) + (shalf << 2);
    gx[0] = xp[0]; gx[1] = xp[1]; gx[2] = xp[2]; gx[3] = xp[3];
    const v4i* qp = (const v4i*)(qbase + (kb >> 1)) + (shalf << 1);
    gq[0] = qp[0]; gq[1] = qp[1];
    const int g = gbase + (kb >> 7);              // group constant within this K-slice
    gsc  = scales[g];
    gnzs = -(zeros[g] + 1024.0f) * gsc;           // folds the 0x44800000 bias below
    __builtin_prefetch(xbase + kb + 2 * BK, 0, 0);           // global_prefetch_b8
    __builtin_prefetch(qbase + ((kb + 2 * BK) >> 1), 0, 0);
  };

  auto STORES = [&](int b) {
    // x: fp32 -> bf16
    unsigned int px[8];
#pragma unroll
    for (int j = 0; j < 4; ++j) {
      px[2 * j]     = pk_bf16(gx[j][0], gx[j][1]);
      px[2 * j + 1] = pk_bf16(gx[j][2], gx[j][3]);
    }
    v4u* da = (v4u*)&sA[b][srow * LDSS + (shalf << 4)];
    v4u ta0 = {px[0], px[1], px[2], px[3]};
    v4u ta1 = {px[4], px[5], px[6], px[7]};
    da[0] = ta0; da[1] = ta1;

    // W: int4 dequant -> bf16.  flat f = o*IN + k; even f = hi nibble, odd = lo nibble.
    // Integer-domain nibble->float: asfloat(0x44800000 | n) == 1024.0f + n/8192 (exact),
    // so fma(asfloat(0x44800000|n), s, -(z+1024)*s) == (n/8192 - z)*s.  No v_cvt needed.
    unsigned int pw[8];
#pragma unroll
    for (int u = 0; u < 2; ++u)
#pragma unroll
      for (int j = 0; j < 4; ++j) {
        const unsigned int q  = (unsigned int)gq[u][j];
        const unsigned int qh = 0x44800000u | ((q >> 4) & 15u);   // k even
        const unsigned int ql = 0x44800000u | (q & 15u);          // k odd
        const float whi = fmaf(__uint_as_float(qh), gsc, gnzs);
        const float wlo = fmaf(__uint_as_float(ql), gsc, gnzs);
        pw[u * 4 + j] = pk_bf16(whi, wlo);
      }
    v4u* db = (v4u*)&sB[b][srow * LDSS + (shalf << 4)];
    v4u tb0 = {pw[0], pw[1], pw[2], pw[3]};
    v4u tb1 = {pw[4], pw[5], pw[6], pw[7]};
    db[0] = tb0; db[1] = tb1;
  };

  v8f acc[4][2] = {};
  const int r15 = lane & 15;
  const int hs  = lane >> 4;

  auto COMPUTE = [&](int b) {
    Frag fa[4], fb[2];
    // A frag (16x32 bf16): lanes 0-15 row=r15 K{0..7,16..23}; lanes 16-31 K{8..15,24..31}
#pragma unroll
    for (int mt = 0; mt < 4; ++mt) {
      const unsigned short* p =
          &sA[b][(wave_m * 64 + mt * 16 + r15) * LDSS + (hs << 3)];
      fa[mt].q[0] = *(const v4u*)p;
      fa[mt].q[1] = *(const v4u*)(p + 16);
    }
    // B frag (32x16 bf16): lane n = column; lanes 0-15 K{0..7,16..23}; lanes 16-31 K{8..15,24..31}
#pragma unroll
    for (int nt = 0; nt < 2; ++nt) {
      const unsigned short* p =
          &sB[b][(wave_n * 32 + nt * 16 + r15) * LDSS + (hs << 3)];
      fb[nt].q[0] = *(const v4u*)p;
      fb[nt].q[1] = *(const v4u*)(p + 16);
    }
#pragma unroll
    for (int mt = 0; mt < 4; ++mt)
#pragma unroll
      for (int nt = 0; nt < 2; ++nt)
        acc[mt][nt] = __builtin_amdgcn_wmma_f32_16x16x32_bf16(
            false, fa[mt].v, false, fb[nt].v, (short)0, acc[mt][nt], false, false);
  };

  LOADG(0);
  STORES(0);
  for (int s = 0; s < NKS; ++s) {
    if (s + 1 < NKS) LOADG((s + 1) * BK);   // next global fetch in flight over compute
    __syncthreads();                         // stage s visible to all waves
    COMPUTE(s & 1);
    if (s + 1 < NKS) STORES((s + 1) & 1);   // writes other buffer; safe w.r.t. current reads
  }

  // Epilogue: C layout — VGPR r holds row r (lanes 0-15) / r+8 (lanes 16-31), lane&15 = column
#pragma unroll
  for (int nt = 0; nt < 2; ++nt) {
    const int col  = blockIdx.x * BN + wave_n * 32 + nt * 16 + r15;
    const float bv = bias[col];
#pragma unroll
    for (int mt = 0; mt < 4; ++mt) {
      const size_t rowg = (size_t)blockIdx.y * BM + wave_m * 64 + mt * 16 + (hs << 3);
#pragma unroll
      for (int rr = 0; rr < 8; ++rr) {
        out[(rowg + rr) * OUT_F + col] = acc[mt][nt][rr] + bv;
      }
    }
  }
}

extern "C" void kernel_launch(void* const* d_in, const int* in_sizes, int n_in,
                              void* d_out, int out_size, void* d_ws, size_t ws_size,
                              hipStream_t stream) {
  const float* x      = (const float*)d_in[0];
  const int*   qw     = (const int*)d_in[1];
  const float* zeros  = (const float*)d_in[2];
  const float* scales = (const float*)d_in[3];
  const float* bias   = (const float*)d_in[4];
  float*       out    = (float*)d_out;

  const int tokens = in_sizes[0] / IN_F;      // 8192
  dim3 grid(OUT_F / BN, tokens / BM);         // 86 x 64
  dim3 block(256);                            // 8 wave32 waves
  QuantLinearA16W4_57174604644898_kernel<<<grid, block, 0, stream>>>(
      x, qw, zeros, scales, bias, out);
}
